// TensorTrainProjection_64587718197486
// MI455X (gfx1250) — compile-verified, matrix operand order checked by reading, not run on hardware
//
#include <hip/hip_runtime.h>
#include <hip/hip_bf16.h>
#include <stdint.h>

// CDNA5 WMMA operand types
typedef __attribute__((ext_vector_type(16))) __bf16 v16bf;
typedef __attribute__((ext_vector_type(8)))  float  v8f;

union AOP { uint32_t u[8]; uint4 q[2]; v16bf v; };

// LDS intermediate buffers (bf16), padded rows for 16B alignment + bank stagger:
//   shape 512 x 64  -> row stride 72 elems (144B = 9*16B, stagger 36 banks)
//   shape 1024 x 32 -> row stride 40 elems ( 80B = 5*16B, stagger 20 banks)
#define STR64 72
#define STR32 40
#define BUF0_ELEMS 40960   // max(512*72, 1024*40)
#define BUF1_ELEMS 36864   // 512*72

__device__ __host__ inline uint32_t pack_bf16(float lo, float hi) {
  union { __bf16 b; uint16_t s; } a, c;
  a.b = (__bf16)lo; c.b = (__bf16)hi;
  return ((uint32_t)c.s << 16) | (uint32_t)a.s;
}

// B-operand element mapping for v_wmma_*_16x16x32 (mirror of the documented A
// layout): element (K,N): lane = N%16 + 16*((K>>3)&1), vgpr = ((K&7)>>1)|((K>>4)<<2),
// half = K&1.  Inverse used by prep:
__device__ inline int kmap(int v, int lane, int h) {
  return ((v >> 2) << 4) | ((lane >> 4) << 3) | ((v & 3) << 1) | h;
}

__device__ inline v8f wmma_bf16(const AOP& a, const AOP& b, v8f c) {
  return __builtin_amdgcn_wmma_f32_16x16x32_bf16(false, a.v, false, b.v, (short)0, c, false, false);
}

__device__ inline void load_B(AOP& b, const uint32_t* __restrict__ wl, int tileIdx, int lane) {
  const uint4* p = (const uint4*)(wl + (((tileIdx * 32) + lane) << 3));
  b.q[0] = p[0]; b.q[1] = p[1];
}

// ---------------------------------------------------------------------------
// Prep: contract adjacent TT cores over their shared 16-wide bond and pack the
// fused weights into per-lane WMMA B-operand layout (u32 WL[tile][lane][vgpr]).
//   wl01 : (cores 0*1)  32x256 (K rows 16..31 zero)  -> 16 tiles
//   wl23 : (cores 2*3)  64x64                        -> 4 col tiles x 2 K-halves
//   wl45 : (cores 4*5)  64x64                        -> 4 x 2
//   wl6  : core 6       32x32                        -> 2 tiles
//   wl78 : (cores 7*8)  64x16 (cols 4..15 zero)      -> 1 col tile x 2 K-halves
// core 7 has swapped bond dims in the reference (g7[a][b][l][r] = p7[a][b][r][l]).
// ---------------------------------------------------------------------------
__global__ void tt_prep(const float* __restrict__ p0_, const float* __restrict__ p1_,
                        const float* __restrict__ p2_, const float* __restrict__ p3_,
                        const float* __restrict__ p4_, const float* __restrict__ p5_,
                        const float* __restrict__ p6_, const float* __restrict__ p7_,
                        const float* __restrict__ p8_,
                        uint32_t* __restrict__ wl01, uint32_t* __restrict__ wl23,
                        uint32_t* __restrict__ wl45, uint32_t* __restrict__ wl6,
                        uint32_t* __restrict__ wl78) {
  const int tid = threadIdx.x;

  // wl01: W[k=(a0*2+a1)][n=(b0*2+b1)*16+r] = sum_r0 p0[a0,b0,0,r0] * p1[a1,b1,r0,r]
  for (int i = tid; i < 16 * 32 * 8; i += blockDim.x) {
    int v = i & 7, lane = (i >> 3) & 31, tile = i >> 8;
    int n16 = lane & 15;
    float vals[2];
#pragma unroll
    for (int h = 0; h < 2; h++) {
      int k = kmap(v, lane, h);
      float s = 0.f;
      if (k < 16) {
        int a0 = k >> 1, a1 = k & 1, b0 = tile >> 1, b1 = tile & 1;
        for (int r0 = 0; r0 < 16; r0++)
          s += p0_[(a0 * 8 + b0) * 16 + r0] * p1_[((a1 * 2 + b1) * 16 + r0) * 16 + n16];
      }
      vals[h] = s;
    }
    wl01[i] = pack_bf16(vals[0], vals[1]);
  }

  // wl23 / wl45: W[c=(l*4+aA*2+aB)][n=(bA*2+bB)*16+r] = sum_rm A[aA,bA,l,rm]*B[aB,bB,rm,r]
  for (int sec = 0; sec < 2; sec++) {
    const float* pa = sec ? p4_ : p2_;
    const float* pb = sec ? p5_ : p3_;
    uint32_t* wl = sec ? wl45 : wl23;
    for (int i = tid; i < 4 * 2 * 32 * 8; i += blockDim.x) {
      int v = i & 7, lane = (i >> 3) & 31, kh = (i >> 8) & 1, ct = i >> 9;
      int n16 = lane & 15;
      float vals[2];
#pragma unroll
      for (int h = 0; h < 2; h++) {
        int k = kh * 32 + kmap(v, lane, h);
        int l = k >> 2, aA = (k >> 1) & 1, aB = k & 1;
        int bA = ct >> 1, bB = ct & 1;
        float s = 0.f;
        for (int rm = 0; rm < 16; rm++)
          s += pa[((aA * 2 + bA) * 16 + l) * 16 + rm] * pb[((aB * 2 + bB) * 16 + rm) * 16 + n16];
        vals[h] = s;
      }
      wl[i] = pack_bf16(vals[0], vals[1]);
    }
  }

  // wl6: W[c=(l*2+a6)][n=b6*16+r] = p6[a6,b6,l,r]
  for (int i = tid; i < 2 * 32 * 8; i += blockDim.x) {
    int v = i & 7, lane = (i >> 3) & 31, tile = i >> 8;
    int n16 = lane & 15;
    float vals[2];
#pragma unroll
    for (int h = 0; h < 2; h++) {
      int k = kmap(v, lane, h);
      vals[h] = p6_[(((k & 1) * 2 + tile) * 16 + (k >> 1)) * 16 + n16];
    }
    wl6[i] = pack_bf16(vals[0], vals[1]);
  }

  // wl78: W[c=(l*4+a7*2+a8)][n=b7*2+b8 (<4)] = sum_r7 p7[a7,b7,r7,l] * p8[a8,b8,r7]
  for (int i = tid; i < 2 * 32 * 8; i += blockDim.x) {
    int v = i & 7, lane = (i >> 3) & 31, kh = i >> 8;
    int n16 = lane & 15;
    float vals[2];
#pragma unroll
    for (int h = 0; h < 2; h++) {
      int k = kh * 32 + kmap(v, lane, h);
      int l = k >> 2, a7 = (k >> 1) & 1, a8 = k & 1;
      float s = 0.f;
      if (n16 < 4) {
        int b7 = n16 >> 1, b8 = n16 & 1;
        for (int r7 = 0; r7 < 16; r7++)
          s += p7_[((a7 * 2 + b7) * 16 + r7) * 16 + l] * p8_[(a8 * 2 + b8) * 16 + r7];
      }
      vals[h] = s;
    }
    wl78[i] = pack_bf16(vals[0], vals[1]);
  }
}

// Generic fused-pair mid pass: 512x64 @ 64x64. MBITS = #m bits in the row index,
// OSPLIT = how many m bits the NEXT pass consumes, OSTRIDE = output row stride.
template <int MBITS, int OSPLIT, int OSTRIDE>
__device__ inline void fused64(const __bf16* __restrict__ zr, __bf16* __restrict__ zw,
                               const uint32_t* __restrict__ wl,
                               int wave, int lane, int col16, int hi) {
  AOP b[4][2];
#pragma unroll
  for (int ct = 0; ct < 4; ct++) {
    load_B(b[ct][0], wl, ct * 2 + 0, lane);
    load_B(b[ct][1], wl, ct * 2 + 1, lane);
  }
#pragma unroll
  for (int rr = 0; rr < 2; rr++) {
    const int rt = wave + rr * 16;
    AOP alo, ahi;
    const char* base = (const char*)zr + (size_t)(rt * 16 + col16) * (STR64 * 2) + (hi << 4);
    alo.q[0] = *(const uint4*)(base);      alo.q[1] = *(const uint4*)(base + 32);
    ahi.q[0] = *(const uint4*)(base + 64); ahi.q[1] = *(const uint4*)(base + 96);
#pragma unroll
    for (int ct = 0; ct < 4; ct++) {
      v8f c = {};
      c = wmma_bf16(alo, b[ct][0], c);
      c = wmma_bf16(ahi, b[ct][1], c);
#pragma unroll
      for (int v = 0; v < 8; v++) {
        int row = rt * 16 + v + (hi << 3);
        int j  = row >> MBITS;
        int mm = row & ((1 << MBITS) - 1);
        if (OSPLIT == 2) {
          int aa = mm >> (MBITS - 1), ab = (mm >> (MBITS - 2)) & 1;
          int ml = mm & ((1 << (MBITS - 2)) - 1);
          int zrow = ((j * 4 + ct) << (MBITS - 2)) + ml;
          zw[zrow * OSTRIDE + (col16 << 2) + aa * 2 + ab] = (__bf16)c[v];
        } else {
          int aa = mm >> (MBITS - 1);
          int ml = mm & ((1 << (MBITS - 1)) - 1);
          int zrow = ((j * 4 + ct) << (MBITS - 1)) + ml;
          zw[zrow * OSTRIDE + (col16 << 1) + aa] = (__bf16)c[v];
        }
      }
    }
  }
}

// ---------------------------------------------------------------------------
// Fused TT-apply, one workgroup (16 waves) per token, 5 WMMA passes in LDS.
// ---------------------------------------------------------------------------
__global__ void __launch_bounds__(512, 1)
tt_main(const float* __restrict__ x,
        const uint32_t* __restrict__ wl01, const uint32_t* __restrict__ wl23,
        const uint32_t* __restrict__ wl45, const uint32_t* __restrict__ wl6,
        const uint32_t* __restrict__ wl78,
        float* __restrict__ out) {
  extern __shared__ char smem[];
  __bf16* buf0 = (__bf16*)smem;          // P01 out (512x72) / P45 out (1024x40)
  __bf16* buf1 = buf0 + BUF0_ELEMS;      // P23 out / P6 out (512x72)

  const int t     = blockIdx.x;
  const int tid   = threadIdx.x;
  const int wave  = tid >> 5;
  const int lane  = tid & 31;
  const int col16 = lane & 15;
  const int hi    = lane >> 4;

  // ---- P01: x[t] viewed [a0(8)][a1(2)][m(128)]; GEMM 128x16(pad32) @ 32x256
  {
    const int rt     = wave & 7;           // 8 row tiles of m
    const int ctBase = (wave >> 3) << 3;   // 16 col tiles split over wave groups
    const int m      = rt * 16 + col16;
    AOP a;
#pragma unroll
    for (int v = 0; v < 4; v++) {          // K = (hi*4+v)*2 + h  ->  x[a0*256 + a1*128 + m]
      const float* xp = x + (size_t)t * 2048 + (hi * 4 + v) * 256 + m;
      a.u[v]     = pack_bf16(xp[0], xp[128]);
      a.u[v + 4] = 0u;                     // K = 16..31 zero-padded
    }
#pragma unroll
    for (int ci = 0; ci < 8; ci++) {
      const int ct = ctBase + ci;          // ct = j' = b0*2+b1
      AOP b; load_B(b, wl01, ct, lane);
      v8f c = {};
      c = wmma_bf16(a, b, c);
#pragma unroll
      for (int v = 0; v < 8; v++) {        // scatter: m=(a2,a3,m4); col'=(r,a2,a3)
        int mrow = rt * 16 + v + (hi << 3);
        int a2 = mrow >> 6, a3 = (mrow >> 5) & 1, m4 = mrow & 31;
        buf0[(ct * 32 + m4) * STR64 + (col16 << 2) + a2 * 2 + a3] = (__bf16)c[v];
      }
    }
  }
  __syncthreads();

  // ---- P23: 512x64 @ 64x64, rows=(j'(16),m4(32)), next consumes 2 bits
  fused64<5, 2, STR64>(buf0, buf1, wl23, wave, lane, col16, hi);
  __syncthreads();

  // ---- P45: 512x64 @ 64x64, rows=(j''(64),m5(8)), next consumes 1 bit -> 1024x32
  fused64<3, 1, STR32>(buf1, buf0, wl45, wave, lane, col16, hi);
  __syncthreads();

  // ---- P6: 1024x32 @ 32x32 (core 6 alone), next consumes 2 bits -> 512x64
  {
    AOP b0, b1;
    load_B(b0, wl6, 0, lane);
    load_B(b1, wl6, 1, lane);
#pragma unroll
    for (int rr = 0; rr < 4; rr++) {
      const int rt = wave + rr * 16;
      AOP a;
      const char* base = (const char*)buf0 + (size_t)(rt * 16 + col16) * (STR32 * 2) + (hi << 4);
      a.q[0] = *(const uint4*)(base);
      a.q[1] = *(const uint4*)(base + 32);
      v8f c0 = {}, c1 = {};
      c0 = wmma_bf16(a, b0, c0);
      c1 = wmma_bf16(a, b1, c1);
#pragma unroll
      for (int v = 0; v < 8; v++) {        // rows=(j3(256),m6(4)); col'=(r, a7,a8)=col16*4+m6
        int row = rt * 16 + v + (hi << 3);
        int j3 = row >> 2, mm = row & 3;
        int zcol = (col16 << 2) + mm;
        buf1[(j3 * 2 + 0) * STR64 + zcol] = (__bf16)c0[v];
        buf1[(j3 * 2 + 1) * STR64 + zcol] = (__bf16)c1[v];
      }
    }
  }
  __syncthreads();

  // ---- P78: 512x64 @ 64x16 (N=4 valid), results straight to global memory
  {
    AOP blo, bhi;
    load_B(blo, wl78, 0, lane);
    load_B(bhi, wl78, 1, lane);
#pragma unroll
    for (int rr = 0; rr < 2; rr++) {
      const int rt = wave + rr * 16;
      AOP alo, ahi;
      const char* base = (const char*)buf1 + (size_t)(rt * 16 + col16) * (STR64 * 2) + (hi << 4);
      alo.q[0] = *(const uint4*)(base);      alo.q[1] = *(const uint4*)(base + 32);
      ahi.q[0] = *(const uint4*)(base + 64); ahi.q[1] = *(const uint4*)(base + 96);
      v8f c = {};
      c = wmma_bf16(alo, blo, c);
      c = wmma_bf16(ahi, bhi, c);
      if (col16 < 4) {                     // out[t][j4*4 + (b7*2+b8)]
#pragma unroll
        for (int v = 0; v < 8; v++) {
          int row = rt * 16 + v + (hi << 3);
          out[(size_t)t * 2048 + row * 4 + col16] = c[v];
        }
      }
    }
  }
}

extern "C" void kernel_launch(void* const* d_in, const int* in_sizes, int n_in,
                              void* d_out, int out_size, void* d_ws, size_t ws_size,
                              hipStream_t stream) {
  const float* x  = (const float*)d_in[0];
  const float* p0 = (const float*)d_in[1];
  const float* p1 = (const float*)d_in[2];
  const float* p2 = (const float*)d_in[3];
  const float* p3 = (const float*)d_in[4];
  const float* p4 = (const float*)d_in[5];
  const float* p5 = (const float*)d_in[6];
  const float* p6 = (const float*)d_in[7];
  const float* p7 = (const float*)d_in[8];
  const float* p8 = (const float*)d_in[9];
  float* out = (float*)d_out;

  uint32_t* wl01 = (uint32_t*)d_ws;        // 16*32*8 = 4096 u32
  uint32_t* wl23 = wl01 + 4096;            //  4*2*32*8 = 2048 u32
  uint32_t* wl45 = wl23 + 2048;            //  2048 u32
  uint32_t* wl6  = wl45 + 2048;            //  2*32*8 = 512 u32
  uint32_t* wl78 = wl6 + 512;              //  512 u32  (total 36 KB)

  const int T = in_sizes[0] / 2048;        // 8192 tokens
  const size_t smem = (size_t)(BUF0_ELEMS + BUF1_ELEMS) * sizeof(__bf16);  // 152 KB

  (void)hipFuncSetAttribute((const void*)tt_main,
                            hipFuncAttributeMaxDynamicSharedMemorySize, (int)smem);

  tt_prep<<<1, 256, 0, stream>>>(p0, p1, p2, p3, p4, p5, p6, p7, p8,
                                 wl01, wl23, wl45, wl6, wl78);
  tt_main<<<T, 512, smem, stream>>>(x, wl01, wl23, wl45, wl6, wl78, out);
}
